// GRU_38938173505937
// MI455X (gfx1250) — matrix operand order repaired and checked by reading, not compiled, not run on hardware
//
#include <hip/hip_runtime.h>

// ---------------------------------------------------------------------------
// BiGRU (3 layers, H=60) + frame head + time-softmax + batch head, MI455X.
//
//  * gi = x @ W_ih^T + b_ih precomputed for both directions with a WMMA GEMM
//    (f16 A/B, f32 accumulate, f16 store) -> parallel over all T*B rows.
//  * Sequential recurrence: 16 workgroups (2 dirs x 8 batch tiles of 16).
//    W_hh^T WMMA B-fragments live in VGPRs for all 4096 steps; h kept in LDS
//    (f16, safe since |h|<1). Per step: prefetch next step's gi slab
//    (global_prefetch_b8), preload this step's gi into registers BEFORE the
//    barrier (latency overlaps barrier + ds_load_b128 + v_wmma), 6 WMMA per
//    wave, gate math via v_exp/v_rcp, h update, output store.
//  * Frame head fused per time step (M=128 rows = one t): two WMMA stages
//    (120->60 tanh, 60->40 tanh) + VALU dot for the 5-wide logits, scattered
//    into (B,5,T) for the softmax-over-time kernel.
//  * m-head: softmax rows sum to 1 -> inner softmax gives 0.2 per class;
//    compute the (batch-constant) 2-vector analytically.
//
// Workspace layout (bytes), assumed ws_size >= ~630 MB:
//   bufA : T*B*120 f16   (125.8 MB)   layer activations (ping)
//   bufB : T*B*120 f16   (125.8 MB)   layer activations (pong)
//   gi   : T*B*360 f16   (377.5 MB)   per-layer input projections (both dirs)
//   lg   : overlaid on gi region (B*5*T f32) — gi dead when head runs.
// ---------------------------------------------------------------------------

#define TT 4096
#define BB 128
#define HH 60

typedef __attribute__((ext_vector_type(16))) _Float16 v16h;
typedef __attribute__((ext_vector_type(8)))  float    v8f;

union FragU { uint4 q[2]; v16h h; };

// A fragment (16x32 f16, M x K). ISA layout: lanes 0-15 -> M=lane, K base 0;
// lanes 16-31 -> M=lane-16, K base 8; halves 0..7 = K kb..kb+7 (VGPR0..3),
// halves 8..15 = K kb+16..kb+23 (VGPR4..7).
__device__ __forceinline__ v16h ld_frag_A(const _Float16* base, int ldm,
                                          int m0, int k0) {
  const int lane = threadIdx.x & 31;
  const int m  = m0 + (lane & 15);
  const int kb = k0 + ((lane >> 4) << 3);
  const _Float16* p = base + m * ldm + kb;
  FragU u;
  u.q[0] = *(const uint4*)(p);
  u.q[1] = *(const uint4*)(p + 16);
  return u.h;
}

// B fragment (32x16 f16, K x N), element (k,n) = W[n][k] (W row-major n,k).
// Lanes 0-15: n=lane, K=0..7,16..23; lanes 16-31: n=lane-16, K=8..15,24..31.
__device__ __forceinline__ v16h ld_frag_B(const _Float16* base, int ldn,
                                          int n0, int k0) {
  const int lane = threadIdx.x & 31;
  const int n  = n0 + (lane & 15);
  const int kb = k0 + ((lane >> 4) << 3);
  const uint4* p = (const uint4*)(base + n * ldn + kb);
  FragU u;
  u.q[0] = p[0];
  u.q[1] = p[2];
  return u.h;
}

__device__ __forceinline__ v8f wmma32(v16h a, v16h b, v8f c) {
  return __builtin_amdgcn_wmma_f32_16x16x32_f16(false, a, false, b,
                                                (short)0, c, false, false);
}

__device__ __forceinline__ float sigm(float x) {
  return __builtin_amdgcn_rcpf(1.f + __expf(-x));
}
__device__ __forceinline__ float tanh_f(float x) {
  return 1.f - 2.f * __builtin_amdgcn_rcpf(1.f + __expf(2.f * x));
}

// --------------------------- x (B,F,T) -> xh (T,B,F) f16 -------------------
__global__ void __launch_bounds__(256)
k_transpose(const float* __restrict__ x, _Float16* __restrict__ xh) {
  __shared__ float tile[64][65];
  const int b  = blockIdx.x;
  const int t0 = blockIdx.y * 64;
  const float* src = x + (size_t)b * 64 * TT;
  for (int idx = threadIdx.x; idx < 64 * 64; idx += 256) {
    int ii = idx >> 6, tt = idx & 63;
    tile[ii][tt] = src[(size_t)ii * TT + t0 + tt];
  }
  __syncthreads();
  for (int idx = threadIdx.x; idx < 64 * 64; idx += 256) {
    int tt = idx >> 6, ii = idx & 63;
    xh[((size_t)(t0 + tt) * BB + b) * 64 + ii] = (_Float16)tile[ii][tt];
  }
}

// --------- gi = xin @ W^T + b, W = concat(dir0,dir1) rows (360 x In) -------
// Grid: (TT, 6). Block 256 = 8 waves; wave = 1 M-tile (16 rows), 4 N-tiles.
__global__ void __launch_bounds__(256)
k_gi_gemm(const _Float16* __restrict__ xin, int In, int nkt,
          const float* __restrict__ w, const float* __restrict__ bias,
          _Float16* __restrict__ gi) {
  __shared__ __align__(16) _Float16 Alds[128 * 128];
  __shared__ __align__(16) _Float16 Wlds[64 * 128];
  const int tid = threadIdx.x;
  const size_t m0 = (size_t)blockIdx.x * 128;
  const int n0 = blockIdx.y * 64;
  // Stage A tile (128 x In, K-padded to 128 with zeros), u32 granularity.
  for (int idx = tid; idx < 128 * 64; idx += 256) {
    int r = idx >> 6, c2 = (idx & 63) << 1;
    unsigned int v = 0;
    if (c2 < In) {
      const _Float16* g = xin + (m0 + r) * (size_t)In + c2;
      unsigned short a0 = __builtin_bit_cast(unsigned short, g[0]);
      unsigned short a1 = __builtin_bit_cast(unsigned short, g[1]);
      v = (unsigned int)a0 | ((unsigned int)a1 << 16);
    }
    *(unsigned int*)(&Alds[(r << 7) + c2]) = v;
  }
  // Stage W tile (64 x 128, zero padded), f32 -> f16.
  for (int idx = tid; idx < 64 * 128; idx += 256) {
    int r = idx >> 7, c = idx & 127;
    int ng = n0 + r;
    _Float16 v = (_Float16)0.f;
    if (ng < 360 && c < In) v = (_Float16)w[(size_t)ng * In + c];
    Wlds[idx] = v;
  }
  __syncthreads();
  const int wave = tid >> 5, lane = tid & 31;
  v8f acc[4] = {};
  for (int kt = 0; kt < nkt; ++kt) {
    v16h a = ld_frag_A(Alds, 128, wave * 16, kt * 32);
#pragma unroll
    for (int nt = 0; nt < 4; ++nt) {
      v16h b = ld_frag_B(Wlds, 128, nt * 16, kt * 32);
      acc[nt] = wmma32(a, b, acc[nt]);
    }
  }
#pragma unroll
  for (int nt = 0; nt < 4; ++nt) {
    int n = n0 + nt * 16 + (lane & 15);
    if (n < 360) {
      float bv = bias[n];
      size_t mrow = m0 + wave * 16 + ((lane >> 4) << 3);
#pragma unroll
      for (int r = 0; r < 8; ++r)
        gi[(mrow + r) * 360 + n] = (_Float16)(acc[nt][r] + bv);
    }
  }
}

// ------------------------------ GRU recurrence -----------------------------
// Grid: 16 blocks = dir(2) x batch-tile(8 x 16). Block 128 = 4 waves.
// Wave w: N-tiles {3w, 3w+1, 3w+2} of the 180(->192) gate columns, single
// M-tile (16 batch rows). W_hh^T B-fragments live in VGPRs across all steps.
// gi for the current step is preloaded into registers before the barrier and
// the next step's gi slab is prefetched, keeping HBM latency off the serial
// critical path.
__global__ void __launch_bounds__(128)
k_scan(const _Float16* __restrict__ gi,    // (T*B,360)
       const float* __restrict__ w_hh,     // (2,180,60)
       const float* __restrict__ b_hh,     // (2,180)
       _Float16* __restrict__ out) {       // (T*B,120)
  __shared__ __align__(16) _Float16 Wlds[192 * 64];
  __shared__ __align__(16) _Float16 hbuf[16 * 64];
  __shared__ float abuf[16 * 192];
  const int tid = threadIdx.x;
  const int dir = blockIdx.x >> 3;
  const int bbase = (blockIdx.x & 7) * 16;
  const float* wd = w_hh + dir * 180 * HH;
  const float* bd = b_hh + dir * 180;
  for (int idx = tid; idx < 192 * 64; idx += 128) {
    int n = idx >> 6, k = idx & 63;
    _Float16 v = (_Float16)0.f;
    if (n < 180 && k < HH) v = (_Float16)wd[n * HH + k];
    Wlds[idx] = v;
  }
  for (int idx = tid; idx < 16 * 64; idx += 128) hbuf[idx] = (_Float16)0.f;

  const int wave = tid >> 5, lane = tid & 31;
  const int ntbase = 3 * wave;                    // waves 0..3 -> tiles 0..11
  const int ml = (lane >> 4) << 3;                // D rows ml..ml+7
  v16h bf[3][2];
  float bh[3];
  int nn[3];
  bool vmask[3], gmask[3];
#pragma unroll
  for (int i = 0; i < 3; ++i) {
#pragma unroll
    for (int kt = 0; kt < 2; ++kt)
      bf[i][kt] = ld_frag_B(Wlds, 64, (ntbase + i) * 16, kt * 32);
    nn[i] = (ntbase + i) * 16 + (lane & 15);
    vmask[i] = nn[i] < 180;
    gmask[i] = nn[i] < 120;
    bh[i] = vmask[i] ? bd[nn[i]] : 0.f;
  }
  // gate-stage mapping (16*60 = 960 units over 128 threads), loop-invariant
  int gm[8], gj[8];
#pragma unroll
  for (int q = 0; q < 8; ++q) {
    int idx = tid + q * 128;
    gm[q] = idx / HH;
    gj[q] = idx - gm[q] * HH;
  }
  const int goff = dir * 180;

  for (int tt = 0; tt < TT; ++tt) {
    const int t = dir ? (TT - 1 - tt) : tt;
    const size_t rowbase = (size_t)t * BB + bbase;
    // Prefetch next step's contiguous gi slab (16 rows * 720 B = 90 lines).
    if (tt + 1 < TT) {
      const int tn = dir ? (TT - 2 - tt) : (tt + 1);
      const char* pb = (const char*)gi + ((size_t)tn * BB + bbase) * 360 * 2;
      if (tid < 90) __builtin_prefetch(pb + tid * 128, 0, 0);
    }
    // Preload this step's gi into registers (independent of h -> overlaps
    // the barrier, the ds_load_b128 fragment loads and the WMMAs).
    float girz[3][8];
#pragma unroll
    for (int i = 0; i < 3; ++i) {
      if (gmask[i]) {
        const _Float16* gp = gi + (rowbase + ml) * 360 + goff + nn[i];
#pragma unroll
        for (int r = 0; r < 8; ++r) girz[i][r] = (float)gp[(size_t)r * 360];
      } else {
#pragma unroll
        for (int r = 0; r < 8; ++r) girz[i][r] = 0.f;
      }
    }
    _Float16 gin[8];
#pragma unroll
    for (int q = 0; q < 8; ++q) {
      int idx = tid + q * 128;
      gin[q] = (idx < 16 * HH)
                   ? gi[(rowbase + gm[q]) * 360 + goff + 120 + gj[q]]
                   : (_Float16)0.f;
    }
    __syncthreads();  // h ready (prev step's gate writes / init)
    v16h a0 = ld_frag_A(hbuf, 64, 0, 0);
    v16h a1 = ld_frag_A(hbuf, 64, 0, 32);
#pragma unroll
    for (int i = 0; i < 3; ++i) {
      v8f c = {};
      c = wmma32(a0, bf[i][0], c);
      c = wmma32(a1, bf[i][1], c);
      if (vmask[i]) {
#pragma unroll
        for (int r = 0; r < 8; ++r)
          abuf[(ml + r) * 192 + nn[i]] = c[r] + bh[i] + girz[i][r];
      }
    }
    __syncthreads();  // abuf ready
#pragma unroll
    for (int q = 0; q < 8; ++q) {
      int idx = tid + q * 128;
      if (idx < 16 * HH) {
        const int m = gm[q], j = gj[q];
        float ar = abuf[m * 192 + j];
        float az = abuf[m * 192 + 60 + j];
        float ah = abuf[m * 192 + 120 + j];
        float rg = sigm(ar), zg = sigm(az);
        float nc = tanh_f((float)gin[q] + rg * ah);
        float hp = (float)hbuf[m * 64 + j];
        float hn = (1.f - zg) * nc + zg * hp;
        hbuf[m * 64 + j] = (_Float16)hn;
        out[(rowbase + m) * 120 + dir * HH + j] = (_Float16)hn;
      }
    }
  }
}

// ------------------------------- frame head --------------------------------
// One block per time step t (128 rows = all batch). Two WMMA stages + VALU
// dot for the 5-wide logits, scattered into lg (B,5,T).
__global__ void __launch_bounds__(256)
k_head(const _Float16* __restrict__ xin,  // (T*B,120)
       const float* __restrict__ fw1, const float* __restrict__ fb1,
       const float* __restrict__ fw2, const float* __restrict__ fb2,
       const float* __restrict__ fw3, const float* __restrict__ fb3,
       float* __restrict__ lg) {
  __shared__ __align__(16) _Float16 Alds[128 * 128];
  __shared__ __align__(16) _Float16 W1lds[64 * 128];
  __shared__ __align__(16) _Float16 H1[128 * 64];
  __shared__ __align__(16) _Float16 W2lds[48 * 64];
  __shared__ float H2[128 * 48];
  const int tid = threadIdx.x;
  const int t = blockIdx.x;
  const size_t m0 = (size_t)t * 128;
  for (int idx = tid; idx < 128 * 64; idx += 256) {
    int r = idx >> 6, c2 = (idx & 63) << 1;
    unsigned int v = 0;
    if (c2 < 120) {
      const _Float16* g = xin + (m0 + r) * 120 + c2;
      unsigned short a0 = __builtin_bit_cast(unsigned short, g[0]);
      unsigned short a1 = __builtin_bit_cast(unsigned short, g[1]);
      v = (unsigned int)a0 | ((unsigned int)a1 << 16);
    }
    *(unsigned int*)(&Alds[(r << 7) + c2]) = v;
  }
  for (int idx = tid; idx < 64 * 128; idx += 256) {
    int r = idx >> 7, c = idx & 127;
    _Float16 v = (_Float16)0.f;
    if (r < 60 && c < 120) v = (_Float16)fw1[r * 120 + c];
    W1lds[idx] = v;
  }
  for (int idx = tid; idx < 48 * 64; idx += 256) {
    int r = idx >> 6, c = idx & 63;
    _Float16 v = (_Float16)0.f;
    if (r < 40 && c < 60) v = (_Float16)fw2[r * 60 + c];
    W2lds[idx] = v;
  }
  __syncthreads();
  const int wave = tid >> 5, lane = tid & 31;
  {  // stage 1: 120->60 (K=128 padded, N=64 padded), tanh
    v8f acc[4] = {};
#pragma unroll
    for (int kt = 0; kt < 4; ++kt) {
      v16h a = ld_frag_A(Alds, 128, wave * 16, kt * 32);
#pragma unroll
      for (int nt = 0; nt < 4; ++nt) {
        v16h b = ld_frag_B(W1lds, 128, nt * 16, kt * 32);
        acc[nt] = wmma32(a, b, acc[nt]);
      }
    }
#pragma unroll
    for (int nt = 0; nt < 4; ++nt) {
      int n = nt * 16 + (lane & 15);
      float bv = (n < 60) ? fb1[n] : 0.f;
      int ml = wave * 16 + ((lane >> 4) << 3);
#pragma unroll
      for (int r = 0; r < 8; ++r)
        H1[(ml + r) * 64 + n] = (_Float16)tanh_f(acc[nt][r] + bv);
    }
  }
  __syncthreads();
  {  // stage 2: 60->40 (K=64, N=48 padded), tanh
    v8f acc[3] = {};
#pragma unroll
    for (int kt = 0; kt < 2; ++kt) {
      v16h a = ld_frag_A(H1, 64, wave * 16, kt * 32);
#pragma unroll
      for (int nt = 0; nt < 3; ++nt) {
        v16h b = ld_frag_B(W2lds, 64, nt * 16, kt * 32);
        acc[nt] = wmma32(a, b, acc[nt]);
      }
    }
#pragma unroll
    for (int nt = 0; nt < 3; ++nt) {
      int n = nt * 16 + (lane & 15);
      float bv = (n < 40) ? fb2[n] : 0.f;
      int ml = wave * 16 + ((lane >> 4) << 3);
#pragma unroll
      for (int r = 0; r < 8; ++r)
        H2[(ml + r) * 48 + n] = tanh_f(acc[nt][r] + bv);
    }
  }
  __syncthreads();
  // stage 3: 40->5 logits, scatter to (B,5,T)
  for (int idx = tid; idx < 128 * 5; idx += 256) {
    int m = idx / 5, cc = idx - m * 5;  // m == batch index b
    float s = fb3[cc];
    const float* wr = fw3 + cc * 40;
#pragma unroll 8
    for (int k = 0; k < 40; ++k) s += H2[m * 48 + k] * wr[k];
    lg[((size_t)m * 5 + cc) * TT + t] = s;
  }
}

// ---------------------- softmax over time, frame output --------------------
__global__ void __launch_bounds__(256)
k_softmax_t(const float* __restrict__ lg, float* __restrict__ frame) {
  __shared__ float red[256];
  const int row = blockIdx.x;  // b*5 + c, 640 rows
  const float* src = lg + (size_t)row * TT;
  float mx = -1e30f;
  for (int i = threadIdx.x; i < TT; i += 256) mx = fmaxf(mx, src[i]);
  red[threadIdx.x] = mx;
  __syncthreads();
  for (int s = 128; s > 0; s >>= 1) {
    if (threadIdx.x < s)
      red[threadIdx.x] = fmaxf(red[threadIdx.x], red[threadIdx.x + s]);
    __syncthreads();
  }
  mx = red[0];
  __syncthreads();
  float sum = 0.f;
  for (int i = threadIdx.x; i < TT; i += 256) sum += __expf(src[i] - mx);
  red[threadIdx.x] = sum;
  __syncthreads();
  for (int s = 128; s > 0; s >>= 1) {
    if (threadIdx.x < s) red[threadIdx.x] += red[threadIdx.x + s];
    __syncthreads();
  }
  float inv = __builtin_amdgcn_rcpf(red[0]);
  float* dst = frame + (size_t)row * TT;
  for (int i = threadIdx.x; i < TT; i += 256) dst[i] = __expf(src[i] - mx) * inv;
}

// ------------------------------- batch head --------------------------------
// frame rows each sum to exactly 1 (softmax over T) -> softmax over 5 equal
// values == 0.2; the output 2-vector is batch-constant.
__global__ void k_mhead(const float* __restrict__ mw1, const float* __restrict__ mb1,
                        const float* __restrict__ mw2, const float* __restrict__ mb2,
                        float* __restrict__ mout) {
  int b = threadIdx.x;
  if (b >= BB) return;
  float u[5];
#pragma unroll
  for (int c = 0; c < 5; ++c) {
    float s = mb1[c];
#pragma unroll
    for (int k = 0; k < 5; ++k) s += 0.2f * mw1[c * 5 + k];
    u[c] = fmaxf(s, 0.f);
  }
  float o0 = mb2[0], o1 = mb2[1];
#pragma unroll
  for (int c = 0; c < 5; ++c) {
    o0 += u[c] * mw2[c];
    o1 += u[c] * mw2[5 + c];
  }
  float mxo = fmaxf(o0, o1);
  float e0 = __expf(o0 - mxo), e1 = __expf(o1 - mxo);
  float inv = __builtin_amdgcn_rcpf(e0 + e1);
  mout[b * 2 + 0] = e0 * inv;
  mout[b * 2 + 1] = e1 * inv;
}

// ---------------------------------------------------------------------------
extern "C" void kernel_launch(void* const* d_in, const int* in_sizes, int n_in,
                              void* d_out, int out_size, void* d_ws, size_t ws_size,
                              hipStream_t stream) {
  (void)in_sizes; (void)n_in; (void)out_size; (void)ws_size;
  const float* x = (const float*)d_in[0];
  const float* w_ih[3] = {(const float*)d_in[1], (const float*)d_in[5], (const float*)d_in[9]};
  const float* w_hh[3] = {(const float*)d_in[2], (const float*)d_in[6], (const float*)d_in[10]};
  const float* b_ih[3] = {(const float*)d_in[3], (const float*)d_in[7], (const float*)d_in[11]};
  const float* b_hh[3] = {(const float*)d_in[4], (const float*)d_in[8], (const float*)d_in[12]};
  const float* fw1 = (const float*)d_in[13]; const float* fb1 = (const float*)d_in[14];
  const float* fw2 = (const float*)d_in[15]; const float* fb2 = (const float*)d_in[16];
  const float* fw3 = (const float*)d_in[17]; const float* fb3 = (const float*)d_in[18];
  const float* mw1 = (const float*)d_in[19]; const float* mb1 = (const float*)d_in[20];
  const float* mw2 = (const float*)d_in[21]; const float* mb2 = (const float*)d_in[22];

  char* ws = (char*)d_ws;
  const size_t SZ_ACT = (size_t)TT * BB * 120 * sizeof(_Float16);
  _Float16* bufA = (_Float16*)ws;
  _Float16* bufB = (_Float16*)(ws + SZ_ACT);
  _Float16* gi   = (_Float16*)(ws + 2 * SZ_ACT);
  float*    lg   = (float*)(ws + 2 * SZ_ACT);  // overlays gi (dead by then)

  k_transpose<<<dim3(BB, TT / 64), 256, 0, stream>>>(x, bufA);

  _Float16* cur = bufA;
  _Float16* nxt = bufB;
  const int Ins[3] = {64, 120, 120};
  for (int l = 0; l < 3; ++l) {
    const int nkt = (Ins[l] + 31) / 32;
    k_gi_gemm<<<dim3(TT, 6), 256, 0, stream>>>(cur, Ins[l], nkt,
                                               w_ih[l], b_ih[l], gi);
    k_scan<<<16, 128, 0, stream>>>(gi, w_hh[l], b_hh[l], nxt);
    _Float16* tmp = cur; cur = nxt; nxt = tmp;
  }

  k_head<<<TT, 256, 0, stream>>>(cur, fw1, fb1, fw2, fb2, fw3, fb3, lg);
  k_softmax_t<<<BB * 5, 256, 0, stream>>>(lg, (float*)d_out);
  k_mhead<<<1, 128, 0, stream>>>(mw1, mb1, mw2, mb2,
                                 (float*)d_out + (size_t)BB * 5 * TT);
}
